// RPN_46712064311608
// MI455X (gfx1250) — matrix operand order repaired and checked by reading, not compiled, not run on hardware
//
#include <hip/hip_runtime.h>
#include <hip/hip_bf16.h>

typedef __attribute__((ext_vector_type(16))) _Float16 v16h;
typedef __attribute__((ext_vector_type(8)))  _Float16 v8h;
typedef __attribute__((ext_vector_type(8)))  float    v8f;
typedef int v4i __attribute__((vector_size(16)));

#define BS 4
#define CH 512
#define HH 64
#define WW 64
#define AA 9
#define NANCH (HH*WW*AA)      // 36864
#define PRE_NMS 2000
#define NMS_TH 0.7f

// ---- CDNA5 async global->LDS path (guarded: fallback = plain copy) --------
#if defined(__has_builtin)
#  if __has_builtin(__builtin_amdgcn_global_load_async_to_lds_b128) && \
      __has_builtin(__builtin_amdgcn_s_wait_asynccnt)
#    define HAVE_ASYNC_LDS 1
#  endif
#endif
#ifndef HAVE_ASYNC_LDS
#  define HAVE_ASYNC_LDS 0
#endif

#define AS1 __attribute__((address_space(1)))
#define AS3 __attribute__((address_space(3)))

__device__ __forceinline__ void lds_copy16B(const _Float16* g, _Float16* l) {
#if HAVE_ASYNC_LDS
    __builtin_amdgcn_global_load_async_to_lds_b128(
        (AS1 v4i*)g, (AS3 v4i*)l, 0, 0);
#else
    *(v8h*)l = *(const v8h*)g;
#endif
}
__device__ __forceinline__ void lds_copy_wait() {
#if HAVE_ASYNC_LDS
    __builtin_amdgcn_s_wait_asynccnt(0);
#endif
}

// ---------------------------------------------------------------- NCHW f32 -> NHWC f16
__global__ __launch_bounds__(256)
void xt_kernel(const float* __restrict__ x, _Float16* __restrict__ xt) {
    __shared__ _Float16 tile[64][72];            // +8 pad vs bank conflicts
    const int bh = blockIdx.x;                   // b*64 + h
    const int b = bh >> 6, h = bh & 63;
    const int tid = threadIdx.x;
    const float* src = x + (size_t)b * CH * (HH * WW) + (size_t)h * WW;
    _Float16* dst = xt + (((size_t)b * HH + h) * WW) * CH;
    for (int c0 = 0; c0 < CH; c0 += 64) {
        #pragma unroll
        for (int e = tid; e < 64 * 64; e += 256) {      // read: w contiguous
            int cc = e >> 6, w = e & 63;
            tile[cc][w] = (_Float16)src[(size_t)(c0 + cc) * (HH * WW) + w];
        }
        __syncthreads();
        #pragma unroll
        for (int e = tid; e < 64 * 64; e += 256) {      // write: c contiguous
            int w = e >> 6, cc = e & 63;
            dst[(size_t)w * CH + c0 + cc] = tile[cc][w];
        }
        __syncthreads();
    }
}

// conv_w [co][ci][kh][kw] f32 -> Wt [co][kpos][ci] f16
__global__ __launch_bounds__(256)
void wcvt_kernel(const float* __restrict__ w, _Float16* __restrict__ wt) {
    int i = blockIdx.x * 256 + threadIdx.x;
    if (i >= CH * CH * 9) return;
    int co = i / (CH * 9);
    int rem = i % (CH * 9);
    int ci = rem / 9;
    int kp = rem % 9;
    wt[(co * 9 + kp) * CH + ci] = (_Float16)w[i];
}

// ---------------------------------------------------------------- 3x3 conv (WMMA implicit GEMM)
// GEMM: M = co (A = weights), N = w (B = NHWC pixels), K = ci per (kh,kw) tap.
// Block = one (b,h) output row x 64 co; 4 waves in 2x2; 2x2 accumulators each.
#define APITCH 40   // 32 f16 data + pad; 80B rows keep 16B-aligned chunks
__global__ __launch_bounds__(128)
void conv3_wmma_kernel(const _Float16* __restrict__ Xt,   // [b][h][w][ci]
                       const _Float16* __restrict__ Wt,   // [co][kpos][ci]
                       const float*   __restrict__ bias,
                       float*         __restrict__ fout) {
    __shared__ _Float16 As[64 * APITCH];

    const int bh   = blockIdx.x;
    const int b    = bh >> 6;
    const int h    = bh & 63;
    const int co0  = blockIdx.y * 64;
    const int tid  = threadIdx.x;       // 0..127
    const int lane = tid & 31;
    const int wave = tid >> 5;
    const int wco  = (wave >> 1) * 32;  // wave tile origin (M = co)
    const int ww   = (wave & 1) * 32;   // wave tile origin (N = w)

    const int lrow = lane & 15;
    const int kbA  = (lane >> 4) * 16;  // A fragment: contiguous-K layout
    const int kbB  = (lane >> 4) * 16;  // B fragment: contiguous-K layout

    const int arow = tid >> 2;          // staging: 0..31 (and +32)
    const int aseg = (tid & 3) * 8;     // f16 offset (16B segments)

    v8f acc[2][2] = {};

    for (int kpos = 0; kpos < 9; ++kpos) {
        const int kh = kpos / 3, kw = kpos % 3;
        const int hh = h + kh - 1;
        const bool hok = ((unsigned)hh < (unsigned)HH);
        const _Float16* xplane = Xt + (((size_t)b * HH + hh) * WW) * CH;

        for (int ci0 = 0; ci0 < CH; ci0 += 32) {
            // ---- stage 64x32 weight tile into LDS (async b128 per 16B segment)
            lds_copy16B(Wt + ((size_t)(co0 + arow) * 9 + kpos) * CH + ci0 + aseg,
                        &As[arow * APITCH + aseg]);
            lds_copy16B(Wt + ((size_t)(co0 + arow + 32) * 9 + kpos) * CH + ci0 + aseg,
                        &As[(arow + 32) * APITCH + aseg]);
            lds_copy_wait();
            __syncthreads();

            // ---- A fragments from LDS
            v16h afr[2];
            #pragma unroll
            for (int i = 0; i < 2; ++i) {
                const _Float16* pa = &As[(wco + i * 16 + lrow) * APITCH + kbA];
                v8h alo = *(const v8h*)(pa);
                v8h ahi = *(const v8h*)(pa + 8);
                afr[i] = __builtin_shufflevector(alo, ahi,
                          0,1,2,3,4,5,6,7,8,9,10,11,12,13,14,15);
            }
            // ---- B fragments straight from global NHWC (32B per lane, aligned)
            v16h bfr[2];
            #pragma unroll
            for (int j = 0; j < 2; ++j) {
                int ws_ = ww + j * 16 + lrow + kw - 1;
                v16h v = {};
                if (hok && (unsigned)ws_ < (unsigned)WW)
                    v = *(const v16h*)(xplane + (size_t)ws_ * CH + ci0 + kbB);
                bfr[j] = v;
            }
            #pragma unroll
            for (int i = 0; i < 2; ++i)
                #pragma unroll
                for (int j = 0; j < 2; ++j)
                    acc[i][j] = __builtin_amdgcn_wmma_f32_16x16x32_f16(
                        false, afr[i], false, bfr[j], (short)0, acc[i][j],
                        false, false);
            __syncthreads();   // WAR: next stage overwrites As
        }
    }

    // store: per VGPR r -> co = base + r + 8*(lane>>4), w = base + (lane&15)
    #pragma unroll
    for (int i = 0; i < 2; ++i) {
        #pragma unroll
        for (int j = 0; j < 2; ++j) {
            int cob = co0 + wco + i * 16 + ((lane >> 4) << 3);
            int wb  = ww + j * 16 + (lane & 15);
            #pragma unroll
            for (int r = 0; r < 8; ++r) {
                int co = cob + r;
                float v = acc[i][j][r] + bias[co];
                v = fmaxf(v, 0.f);
                fout[(((size_t)b * CH + co) * HH + h) * WW + wb] = v;
            }
        }
    }
}

// ---------------------------------------------------------------- 1x1 heads + anchor decode
__global__ __launch_bounds__(256)
void head_kernel(const float* __restrict__ f,
                 const float* __restrict__ cls_w, const float* __restrict__ cls_b,
                 const float* __restrict__ reg_w, const float* __restrict__ reg_b,
                 const float* __restrict__ anchors,
                 float* __restrict__ scores, float* __restrict__ boxes) {
    int idx = blockIdx.x * 256 + threadIdx.x;   // (b*9 + a)*4096 + pix
    if (idx >= BS * AA * HH * WW) return;
    int pix = idx & 4095;
    int ba  = idx >> 12;
    int a = ba % AA, b = ba / AA;
    int h = pix >> 6, wI = pix & 63;

    const float* fp = f + (size_t)b * CH * (HH * WW) + pix;
    float l0 = cls_b[a * 2], l1 = cls_b[a * 2 + 1];
    float d0 = reg_b[a * 4], d1 = reg_b[a * 4 + 1], d2 = reg_b[a * 4 + 2];
    const float* cw0 = cls_w + (size_t)(a * 2) * CH;
    const float* cw1 = cls_w + (size_t)(a * 2 + 1) * CH;
    const float* rw0 = reg_w + (size_t)(a * 4) * CH;
    const float* rw1 = reg_w + (size_t)(a * 4 + 1) * CH;
    const float* rw2 = reg_w + (size_t)(a * 4 + 2) * CH;
    for (int c = 0; c < CH; ++c) {
        float x = fp[(size_t)c * (HH * WW)];
        l0 += x * cw0[c]; l1 += x * cw1[c];
        d0 += x * rw0[c]; d1 += x * rw1[c]; d2 += x * rw2[c];
    }
    float s  = 1.f / (1.f + expf(l0 - l1));
    float aw = anchors[a * 2], ah = anchors[a * 2 + 1];
    float px = d0 * aw + (float)wI;
    float py = d1 * ah + (float)h;
    float e2 = expf(d2);
    float pw = e2 * aw, ph = e2 * ah;               // reference uses d2 for both
    float x1 = fminf(fmaxf(px - 0.5f * pw, 0.f), (float)(WW - 1));
    float y1 = fminf(fmaxf(py - 0.5f * ph, 0.f), (float)(HH - 1));
    float x2 = fminf(fmaxf(px + 0.5f * pw, 0.f), (float)(WW - 1));
    float y2 = fminf(fmaxf(py + 0.5f * ph, 0.f), (float)(HH - 1));

    int flat = pix * AA + a;
    scores[(size_t)b * NANCH + flat] = s;
    float* bp = boxes + ((size_t)b * NANCH + flat) * 4;
    bp[0] = x1; bp[1] = y1; bp[2] = x2; bp[3] = y2;
}

// ---------------------------------------------------------------- top-2000 (radix select + bitonic sort)
__global__ __launch_bounds__(1024)
void topk_kernel(const float* __restrict__ scores, int* __restrict__ topidx) {
    const int b = blockIdx.x, tid = threadIdx.x;
    const float* sc = scores + (size_t)b * NANCH;
    __shared__ unsigned hist[256];
    __shared__ unsigned sh_prefix, sh_rank, sh_thr, sh_G, cntG, cntE;
    __shared__ unsigned skey[2048];
    __shared__ int      sidx[2048];

    if (tid == 0) { sh_prefix = 0u; sh_rank = PRE_NMS; }
    __syncthreads();
    for (int p = 3; p >= 0; --p) {
        for (int i = tid; i < 256; i += 1024) hist[i] = 0u;
        __syncthreads();
        unsigned pref = sh_prefix;
        unsigned mask = (p == 3) ? 0u : (0xFFFFFFFFu << ((p + 1) * 8));
        for (int i = tid; i < NANCH; i += 1024) {
            unsigned u = __float_as_uint(sc[i]);
            unsigned k = (u & 0x80000000u) ? ~u : (u | 0x80000000u);
            if ((k & mask) == pref)
                atomicAdd(&hist[(k >> (p * 8)) & 255u], 1u);
        }
        __syncthreads();
        if (tid == 0) {
            unsigned r = sh_rank, cum = 0; int chosen = 0;
            for (int bin = 255; bin >= 0; --bin) {
                if (cum + hist[bin] >= r) { chosen = bin; break; }
                cum += hist[bin];
            }
            sh_prefix = pref | ((unsigned)chosen << (p * 8));
            sh_rank = r - cum;
        }
        __syncthreads();
    }
    if (tid == 0) { sh_thr = sh_prefix; sh_G = PRE_NMS - sh_rank; cntG = 0; cntE = 0; }
    __syncthreads();
    for (int i = tid; i < 2048; i += 1024) { skey[i] = 0u; sidx[i] = 0x7fffffff; }
    __syncthreads();
    unsigned T = sh_thr, G = sh_G;
    for (int i = tid; i < NANCH; i += 1024) {
        unsigned u = __float_as_uint(sc[i]);
        unsigned k = (u & 0x80000000u) ? ~u : (u | 0x80000000u);
        if (k > T) {
            unsigned p = atomicAdd(&cntG, 1u);
            skey[p] = k; sidx[p] = i;
        } else if (k == T) {
            unsigned p = atomicAdd(&cntE, 1u);
            if (G + p < PRE_NMS) { skey[G + p] = k; sidx[G + p] = i; }
        }
    }
    __syncthreads();
    for (unsigned size = 2; size <= 2048; size <<= 1) {
        for (unsigned stride = size >> 1; stride > 0; stride >>= 1) {
            for (unsigned t = tid; t < 2048; t += 1024) {
                unsigned ixj = t ^ stride;
                if (ixj > t) {
                    bool desc = ((t & size) == 0);
                    unsigned ka = skey[t], kb = skey[ixj];
                    int ia = sidx[t], ib = sidx[ixj];
                    bool aFirst = (ka > kb) || (ka == kb && ia < ib);
                    if (desc ? !aFirst : aFirst) {
                        skey[t] = kb; skey[ixj] = ka;
                        sidx[t] = ib; sidx[ixj] = ia;
                    }
                }
            }
            __syncthreads();
        }
    }
    for (int i = tid; i < PRE_NMS; i += 1024) topidx[b * PRE_NMS + i] = sidx[i];
}

// ---------------------------------------------------------------- greedy NMS + ordered compaction
__global__ __launch_bounds__(1024)
void nms_kernel(const float* __restrict__ boxes, const int* __restrict__ topidx,
                float* __restrict__ out) {
    const int b = blockIdx.x, tid = threadIdx.x;
    __shared__ float sx1[PRE_NMS], sy1[PRE_NMS], sx2[PRE_NMS], sy2[PRE_NMS], sar[PRE_NMS];
    __shared__ unsigned keep[PRE_NMS];
    __shared__ int ord[PRE_NMS];
    __shared__ int nkeep;

    for (int i = tid; i < PRE_NMS; i += 1024) {
        int g = topidx[b * PRE_NMS + i];
        const float* bp = boxes + ((size_t)b * NANCH + g) * 4;
        float x1 = bp[0], y1 = bp[1], x2 = bp[2], y2 = bp[3];
        sx1[i] = x1; sy1[i] = y1; sx2[i] = x2; sy2[i] = y2;
        sar[i] = (x2 - x1) * (y2 - y1);
        keep[i] = 1u;
    }
    __syncthreads();
    for (int i = 0; i < PRE_NMS; ++i) {
        if (keep[i]) {
            float x1 = sx1[i], y1 = sy1[i], x2 = sx2[i], y2 = sy2[i], ai = sar[i];
            for (int j = i + 1 + tid; j < PRE_NMS; j += 1024) {
                if (!keep[j]) continue;
                float iw = fmaxf(fminf(x2, sx2[j]) - fmaxf(x1, sx1[j]), 0.f);
                float ih = fmaxf(fminf(y2, sy2[j]) - fmaxf(y1, sy1[j]), 0.f);
                float inter = iw * ih;
                float iou = inter / (ai + sar[j] - inter);
                if (iou > NMS_TH) keep[j] = 0u;
            }
        }
        __syncthreads();
    }
    if (tid == 0) {
        int p = 0;
        for (int i = 0; i < PRE_NMS; ++i) if (keep[i]) ord[p++] = i;
        nkeep = p;
    }
    __syncthreads();
    for (int r = tid; r < PRE_NMS; r += 1024) {
        float cx = 0.f, cy = 0.f, cw = 0.f, chh = 0.f;
        if (r < nkeep) {
            int i = ord[r];
            cx = 0.5f * (sx1[i] + sx2[i]); cy = 0.5f * (sy1[i] + sy2[i]);
            cw = sx2[i] - sx1[i];          chh = sy2[i] - sy1[i];
        }
        float* op = out + ((size_t)b * PRE_NMS + r) * 4;
        op[0] = cx; op[1] = cy; op[2] = cw; op[3] = chh;
    }
}

// ---------------------------------------------------------------- launch
extern "C" void kernel_launch(void* const* d_in, const int* in_sizes, int n_in,
                              void* d_out, int out_size, void* d_ws, size_t ws_size,
                              hipStream_t stream) {
    const float* x       = (const float*)d_in[0];
    const float* conv_w  = (const float*)d_in[1];
    const float* conv_b  = (const float*)d_in[2];
    const float* cls_w   = (const float*)d_in[3];
    const float* cls_b   = (const float*)d_in[4];
    const float* reg_w   = (const float*)d_in[5];
    const float* reg_b   = (const float*)d_in[6];
    const float* anchors = (const float*)d_in[7];

    char* ws = (char*)d_ws;
    size_t off = 0;
    auto alloc = [&](size_t bytes) -> void* {
        void* p = ws + off;
        off += (bytes + 255) & ~(size_t)255;
        return p;
    };
    const size_t nX = (size_t)BS * CH * HH * WW;
    _Float16* Xt   = (_Float16*)alloc(nX * 2);                 // NHWC f16
    _Float16* Wt   = (_Float16*)alloc((size_t)CH * 9 * CH * 2);
    float* fbuf    = (float*)alloc(nX * 4);
    float* scoresW = (float*)alloc((size_t)BS * NANCH * 4);
    float* boxesW  = (float*)alloc((size_t)BS * NANCH * 4 * 4);
    int* topidx    = (int*)alloc((size_t)BS * PRE_NMS * 4);

    xt_kernel<<<BS * HH, 256, 0, stream>>>(x, Xt);
    wcvt_kernel<<<(CH * CH * 9 + 255) / 256, 256, 0, stream>>>(conv_w, Wt);
    conv3_wmma_kernel<<<dim3(BS * HH, CH / 64), 128, 0, stream>>>(Xt, Wt, conv_b, fbuf);
    head_kernel<<<(BS * AA * HH * WW + 255) / 256, 256, 0, stream>>>(
        fbuf, cls_w, cls_b, reg_w, reg_b, anchors, scoresW, boxesW);
    topk_kernel<<<BS, 1024, 0, stream>>>(scoresW, topidx);
    nms_kernel<<<BS, 1024, 0, stream>>>(boxesW, topidx, (float*)d_out);
}
